// ResNet_58763742544549
// MI455X (gfx1250) — matrix operand order repaired and checked by reading, not compiled, not run on hardware
//
#include <hip/hip_runtime.h>

typedef _Float16 half_t;
typedef __attribute__((ext_vector_type(4)))  _Float16 v4h;
typedef __attribute__((ext_vector_type(8)))  _Float16 v8h;
typedef __attribute__((ext_vector_type(16))) _Float16 v16h;
typedef __attribute__((ext_vector_type(4)))  float    v4f;
typedef __attribute__((ext_vector_type(8)))  float    v8f;

#define WAVES_PER_BLOCK 4
#define BLOCK_THREADS   (WAVES_PER_BLOCK * 32)

// ---------------------------------------------------------------------------
// Per-wave LDS: im2col panels, row-major [row = t*16 + sample][K + 8 pad].
// ---------------------------------------------------------------------------
#define RS_XIN 8
#define RS_TOQ 40
#define RS_DB  72
#define RS_DR  104
#define RS_QB  72
#define RS_QR  104
#define RS_PV  104

#define REG_A_H 3840
#define REG_B_H 6912
#define REG_C_H 9984
#define PV_OFF  (3 * 16 * RS_QB)
#define WAVE_H  (REG_A_H + REG_B_H + REG_C_H)   // 20736 halfs = 41472 B

// f16 weight staging in d_ws (halfs): rows [cout][K] contiguous.
#define WOFF_TOD 0
#define WOFF_DB   (WOFF_TOD + 32 * 4)
#define WOFF_DR1  (WOFF_DB  + 32 * 64)
#define WOFF_DR2  (WOFF_DR1 + 32 * 96)
#define WOFF_DR3  (WOFF_DR2 + 32 * 96)
#define WOFF_TOQ  (WOFF_DR3 + 32 * 96)
#define WOFF_QB   (WOFF_TOQ + 32 * 32)
#define WOFF_QR1  (WOFF_QB  + 32 * 64)
#define WOFF_QR2  (WOFF_QR1 + 32 * 96)
#define WOFF_QR3  (WOFF_QR2 + 32 * 96)
#define WOFF_VS   (WOFF_QR3 + 32 * 96)
#define WTOTAL_H  (WOFF_VS  + 64 * 96)          // 29824 halfs = 59648 B

// ---------------------------------------------------------------------------
static __device__ __forceinline__ v8f wmma_f16(v16h a, v16h b, v8f c) {
  return __builtin_amdgcn_wmma_f32_16x16x32_f16(
      false, a, false, b, (short)0, c, false, false);
}

// A fragment: two ds_load_b128 from an im2col panel row (offsets all-immediate
// when callers pass compile-time t/kb off a hoisted lane base).
static __device__ __forceinline__ v16h load_A_at(const half_t* laneBase, int off) {
  v8h lo = *(const v8h*)(laneBase + off);
  v8h hh = *(const v8h*)(laneBase + off + 16);
  v16h a;
#pragma unroll
  for (int i = 0; i < 8; ++i) { a[i] = lo[i]; a[8 + i] = hh[i]; }
  return a;
}

// B fragment from staged f16 weights [cout][K]: two aligned 16B loads (k = 8h.., 16+8h..).
static __device__ __forceinline__ v16h load_Bf(const half_t* __restrict__ wf, int K,
                                               int tileN, int kb, int lane) {
  int col = tileN * 16 + (lane & 15);
  int hi  = lane >> 4;
  const half_t* p = wf + col * K + kb + hi * 8;
  v8h lo = *(const v8h*)(p);
  v8h hh = *(const v8h*)(p + 16);
  v16h b;
#pragma unroll
  for (int i = 0; i < 8; ++i) { b[i] = lo[i]; b[8 + i] = hh[i]; }
  return b;
}

static __device__ __forceinline__ v8f bias_splat(const float* __restrict__ bias, int col) {
  float b = bias[col];
  v8f c;
#pragma unroll
  for (int i = 0; i < 8; ++i) c[i] = b;
  return c;
}

// Generic conv layer, fully unrolled; bias folded into WMMA C accumulator.
template <int CHUNKS, int TOUT, int INRS, int OUTRS, int OUTMUL, int OUTADD, bool RELU>
static __device__ __forceinline__ void conv_panel(const half_t* inP, half_t* outP,
                                                  const half_t* __restrict__ wf,
                                                  const float* __restrict__ bias,
                                                  int lane) {
  const int nIdx = lane & 15, hi = lane >> 4;
  v16h B[2][CHUNKS];
#pragma unroll
  for (int tt = 0; tt < 2; ++tt)
#pragma unroll
    for (int c = 0; c < CHUNKS; ++c) B[tt][c] = load_Bf(wf, 32 * CHUNKS, tt, 32 * c, lane);
  v8f C0[2];
#pragma unroll
  for (int tt = 0; tt < 2; ++tt) C0[tt] = bias_splat(bias, tt * 16 + nIdx);

  const half_t* aB = inP + nIdx * INRS + hi * 8;
  half_t* oB = outP + (hi * 8) * OUTRS + nIdx * OUTMUL + OUTADD;
#pragma unroll
  for (int t = 0; t < TOUT; ++t) {
    v16h A[CHUNKS];
#pragma unroll
    for (int c = 0; c < CHUNKS; ++c) A[c] = load_A_at(aB, t * 16 * INRS + 32 * c);
#pragma unroll
    for (int tt = 0; tt < 2; ++tt) {
      v8f acc = C0[tt];
#pragma unroll
      for (int c = 0; c < CHUNKS; ++c) acc = wmma_f16(A[c], B[tt][c], acc);
      half_t* o = oB + tt * 16 * OUTMUL + t * 16 * OUTRS;
#pragma unroll
      for (int i = 0; i < 8; ++i) {
        float v = acc[i];
        if (RELU) v = fmaxf(v, 0.0f);
        o[i * OUTRS] = (half_t)v;
      }
    }
  }
}

// ---------------------------------------------------------------------------
__global__ void prep_weights(const float* w_tod, const float* w_db,
                             const float* w_dr1, const float* w_dr2, const float* w_dr3,
                             const float* w_toq, const float* w_qb,
                             const float* w_qr1, const float* w_qr2, const float* w_qr3,
                             const float* w_vs, half_t* __restrict__ wf) {
  const float* srcs[11] = {w_tod, w_db, w_dr1, w_dr2, w_dr3, w_toq,
                           w_qb, w_qr1, w_qr2, w_qr3, w_vs};
  const int offs[12] = {WOFF_TOD, WOFF_DB, WOFF_DR1, WOFF_DR2, WOFF_DR3, WOFF_TOQ,
                        WOFF_QB, WOFF_QR1, WOFF_QR2, WOFF_QR3, WOFF_VS, WTOTAL_H};
  int tid = blockIdx.x * blockDim.x + threadIdx.x;
  int stride = gridDim.x * blockDim.x;
  for (int a = 0; a < 11; ++a) {
    int n = offs[a + 1] - offs[a];
    for (int i = tid; i < n; i += stride) wf[offs[a] + i] = (half_t)srcs[a][i];
  }
}

// ---------------------------------------------------------------------------
__global__ __launch_bounds__(BLOCK_THREADS)
void resnet_fused_wmma(
    const float* __restrict__ x, const half_t* __restrict__ wf,
    const float* b_tod, const float* b_db,
    const float* b_dr1, const float* b_dr2, const float* b_dr3,
    const float* b_toq, const float* b_qb,
    const float* b_qr1, const float* b_qr2, const float* b_qr3,
    const float* b_vs,  const float* w_out, const float* b_out,
    float* __restrict__ out, int nTotal) {
  extern __shared__ half_t lds[];

  int lane = threadIdx.x & 31;
  int nIdx = lane & 15;
  int hi   = lane >> 4;
  int widx = threadIdx.x >> 5;
  int n0   = (blockIdx.x * WAVES_PER_BLOCK + widx) * 16;
  if (n0 >= nTotal) return;

  half_t* regA = lds + widx * WAVE_H;
  half_t* regB = regA + REG_A_H;
  half_t* regC = regB + REG_B_H;
  half_t* XIN  = regA;
  half_t* PTOQ = regA;
  half_t* PDB  = regB;
  half_t* PQB  = regB;
  half_t* PV   = regB + PV_OFF;
  half_t* PDR  = regC;
  half_t* PQR  = regC;

  // ---- load x: each lane reads a contiguous 24-float run (6x b128);
  //      all scatter indices compile-time off one lane base ------------------
  {
    const float* xg = x + (size_t)n0 * 48 + lane * 24;   // lane covers half a sample
    half_t* xB = XIN + (lane >> 1) * RS_XIN + (lane & 1) * 2;   // row n=lane/2, cin base
#pragma unroll
    for (int k = 0; k < 24; k += 4) {
      v4f f = *(const v4f*)(xg + k);
#pragma unroll
      for (int e = 0; e < 4; ++e) {
        const int kk = k + e;                  // 0..23, compile-time
        const int cin = kk / 12, p = kk % 12;  // compile-time
        xB[p * 16 * RS_XIN + cin] = (half_t)f[e];
      }
    }
  }

  // ---- toDijet: 4 -> 32 channels, K=1 conv (K=4 GEMM, zero-padded) ---------
  {
    v16h B[2];
#pragma unroll
    for (int tt = 0; tt < 2; ++tt) {
      v4h w4 = *(const v4h*)(wf + WOFF_TOD + (tt * 16 + nIdx) * 4);
      v16h b;
#pragma unroll
      for (int s = 0; s < 16; ++s)
        b[s] = (hi == 0 && s < 4) ? w4[s & 3] : (half_t)0.0f;
      B[tt] = b;
    }
    v8f C0[2];
#pragma unroll
    for (int tt = 0; tt < 2; ++tt) C0[tt] = bias_splat(b_tod, tt * 16 + nIdx);

    const half_t* aB = XIN + nIdx * RS_XIN;
    half_t* oDB = PDB + (hi * 8) * RS_DB + nIdx * 2;
    half_t* oDR = PDR + (hi * 8) * RS_DR + nIdx * 3;
#pragma unroll
    for (int p = 0; p < 12; ++p) {
      v4h x4 = *(const v4h*)(aB + p * 16 * RS_XIN);      // ds_load_b64
      v16h A;
#pragma unroll
      for (int s = 0; s < 16; ++s)
        A[s] = (s < 4 && hi == 0) ? x4[s & 3] : (half_t)0.0f;
      const int t = p >> 1, r = p & 1;
#pragma unroll
      for (int tt = 0; tt < 2; ++tt) {
        v8f acc = C0[tt];
        acc = wmma_f16(A, B[tt], acc);
        half_t* o1 = oDB + tt * 32 + t * 16 * RS_DB + r;
        half_t* o2 = oDR + tt * 48 + t * 16 * RS_DR + r;
#pragma unroll
        for (int i = 0; i < 8; ++i) {
          half_t v = (half_t)acc[i];
          o1[i * RS_DB] = v;
          o2[i * RS_DR] = v;
        }
      }
    }
  }

  // ---- dijet stage ---------------------------------------------------------
  conv_panel<2, 6, RS_DB, RS_DR,  3, 2, true >(PDB, PDR,  wf + WOFF_DB,  b_db,  lane);
  conv_panel<3, 6, RS_DR, RS_DR,  3, 2, true >(PDR, PDR,  wf + WOFF_DR1, b_dr1, lane);
  conv_panel<3, 6, RS_DR, RS_DR,  3, 2, true >(PDR, PDR,  wf + WOFF_DR2, b_dr2, lane);
  conv_panel<3, 6, RS_DR, RS_TOQ, 1, 0, false>(PDR, PTOQ, wf + WOFF_DR3, b_dr3, lane);

  // ---- toQuadjet: 32 -> 32, K=1 conv; dual-store into qb + qr panels -------
  {
    v16h B[2];
#pragma unroll
    for (int tt = 0; tt < 2; ++tt) B[tt] = load_Bf(wf + WOFF_TOQ, 32, tt, 0, lane);
    v8f C0[2];
#pragma unroll
    for (int tt = 0; tt < 2; ++tt) C0[tt] = bias_splat(b_toq, tt * 16 + nIdx);

    const half_t* aB = PTOQ + nIdx * RS_TOQ + hi * 8;
    half_t* oQB = PQB + (hi * 8) * RS_QB + nIdx * 2;
    half_t* oQR = PQR + (hi * 8) * RS_QR + nIdx * 3;
#pragma unroll
    for (int p = 0; p < 6; ++p) {
      v16h A = load_A_at(aB, p * 16 * RS_TOQ);
      const int t = p >> 1, r = p & 1;
#pragma unroll
      for (int tt = 0; tt < 2; ++tt) {
        v8f acc = C0[tt];
        acc = wmma_f16(A, B[tt], acc);
        half_t* o1 = oQB + tt * 32 + t * 16 * RS_QB + r;
        half_t* o2 = oQR + tt * 48 + t * 16 * RS_QR + r;
#pragma unroll
        for (int i = 0; i < 8; ++i) {
          half_t v = (half_t)acc[i];
          o1[i * RS_QB] = v;
          o2[i * RS_QR] = v;
        }
      }
    }
  }

  // ---- quadjetBuilder (stride-2, relu); also saves q1 into PV --------------
  {
    v16h B[2][2];
#pragma unroll
    for (int tt = 0; tt < 2; ++tt)
#pragma unroll
      for (int c = 0; c < 2; ++c) B[tt][c] = load_Bf(wf + WOFF_QB, 64, tt, 32 * c, lane);
    v8f C0[2];
#pragma unroll
    for (int tt = 0; tt < 2; ++tt) C0[tt] = bias_splat(b_qb, tt * 16 + nIdx);

    const half_t* aB = PQB + nIdx * RS_QB + hi * 8;
    half_t* oQR = PQR + (hi * 8) * RS_QR + nIdx * 3 + 2;
    half_t* oPV = PV  + (hi * 8) * RS_PV + nIdx * 3;
#pragma unroll
    for (int t = 0; t < 3; ++t) {
      v16h A0 = load_A_at(aB, t * 16 * RS_QB);
      v16h A1 = load_A_at(aB, t * 16 * RS_QB + 32);
#pragma unroll
      for (int tt = 0; tt < 2; ++tt) {
        v8f acc = C0[tt];
        acc = wmma_f16(A0, B[tt][0], acc);
        acc = wmma_f16(A1, B[tt][1], acc);
        half_t* o1 = oQR + tt * 48 + t * 16 * RS_QR;
        half_t* o2 = oPV + tt * 48 + t;
#pragma unroll
        for (int i = 0; i < 8; ++i) {
          half_t v = (half_t)fmaxf(acc[i], 0.0f);
          o1[i * RS_QR] = v;           // q -> qr d-slot
          o2[i * RS_PV] = v;           // q1 -> vs panel
        }
      }
    }
  }

  // ---- quadjetResNet 1,2 (relu, in-place d-slots) --------------------------
  conv_panel<3, 3, RS_QR, RS_QR, 3, 2, true>(PQR, PQR, wf + WOFF_QR1, b_qr1, lane);
  conv_panel<3, 3, RS_QR, RS_QR, 3, 2, true>(PQR, PQR, wf + WOFF_QR2, b_qr2, lane);

  // ---- quadjetResNet 3 (no relu) fused with v = relu(q + q1) into PV -------
  {
    v16h B[2][3];
#pragma unroll
    for (int tt = 0; tt < 2; ++tt)
#pragma unroll
      for (int c = 0; c < 3; ++c) B[tt][c] = load_Bf(wf + WOFF_QR3, 96, tt, 32 * c, lane);
    v8f C0[2];
#pragma unroll
    for (int tt = 0; tt < 2; ++tt) C0[tt] = bias_splat(b_qr3, tt * 16 + nIdx);

    const half_t* aB = PQR + nIdx * RS_QR + hi * 8;
    half_t* oPV = PV + (hi * 8) * RS_PV + nIdx * 3;
#pragma unroll
    for (int t = 0; t < 3; ++t) {
      v16h A0 = load_A_at(aB, t * 16 * RS_QR);
      v16h A1 = load_A_at(aB, t * 16 * RS_QR + 32);
      v16h A2 = load_A_at(aB, t * 16 * RS_QR + 64);
#pragma unroll
      for (int tt = 0; tt < 2; ++tt) {
        v8f acc = C0[tt];
        acc = wmma_f16(A0, B[tt][0], acc);
        acc = wmma_f16(A1, B[tt][1], acc);
        acc = wmma_f16(A2, B[tt][2], acc);
        half_t* o = oPV + tt * 48 + t;
#pragma unroll
        for (int i = 0; i < 8; ++i) {                // RMW: v = relu(q3 + q1)
          float v = acc[i] + (float)o[i * RS_PV];
          o[i * RS_PV] = (half_t)fmaxf(v, 0.0f);
        }
      }
    }
  }

  // ---- vs conv (96 -> 64, one position) + final 64->1 linear, fused --------
  const half_t* aV = PV + nIdx * RS_PV + hi * 8;
  v16h A0 = load_A_at(aV, 0);
  v16h A1 = load_A_at(aV, 32);
  v16h A2 = load_A_at(aV, 64);

  float p[8];
#pragma unroll
  for (int v = 0; v < 8; ++v) p[v] = 0.0f;

#pragma unroll
  for (int tt = 0; tt < 4; ++tt) {
    v16h B0 = load_Bf(wf + WOFF_VS, 96, tt, 0,  lane);
    v16h B1 = load_Bf(wf + WOFF_VS, 96, tt, 32, lane);
    v16h B2 = load_Bf(wf + WOFF_VS, 96, tt, 64, lane);
    v8f acc = bias_splat(b_vs, tt * 16 + nIdx);
    acc = wmma_f16(A0, B0, acc);
    acc = wmma_f16(A1, B1, acc);
    acc = wmma_f16(A2, B2, acc);
    float wo = w_out[tt * 16 + nIdx];
#pragma unroll
    for (int v = 0; v < 8; ++v)
      p[v] += fmaxf(acc[v], 0.0f) * wo;
  }

#pragma unroll
  for (int m = 1; m < 16; m <<= 1)
#pragma unroll
    for (int v = 0; v < 8; ++v) p[v] += __shfl_xor(p[v], m, 32);

  float b0 = b_out[0];
#pragma unroll
  for (int v = 0; v < 8; ++v)
    if (nIdx == v) out[n0 + hi * 8 + v] = p[v] + b0;
}

// ---------------------------------------------------------------------------
extern "C" void kernel_launch(void* const* d_in, const int* in_sizes, int n_in,
                              void* d_out, int out_size, void* d_ws, size_t ws_size,
                              hipStream_t stream) {
  (void)n_in; (void)ws_size; (void)out_size;
  const float* p[25];
  for (int i = 0; i < 25; ++i) p[i] = (const float*)d_in[i];
  half_t* wf = (half_t*)d_ws;   // needs WTOTAL_H*2 = 59648 B of scratch

  // stage 1: weights f32 -> f16 (once per launch, deterministic)
  prep_weights<<<32, 256, 0, stream>>>(
      p[1], p[3], p[5], p[7], p[9], p[11], p[13], p[15], p[17], p[19], p[21], wf);

  int nTotal = in_sizes[0] / 48;
  int waves  = (nTotal + 15) / 16;
  int blocks = (waves + WAVES_PER_BLOCK - 1) / WAVES_PER_BLOCK;
  size_t shmem = (size_t)WAVES_PER_BLOCK * WAVE_H * sizeof(half_t);  // 165888 B
  resnet_fused_wmma<<<blocks, BLOCK_THREADS, shmem, stream>>>(
      p[0], wf,
      p[2],  p[4],  p[6],  p[8],  p[10], p[12], p[14],
      p[16], p[18], p[20], p[22], p[23], p[24],
      (float*)d_out, nTotal);
}
